// EdgeContactProcessorContactModule_87608742903959
// MI455X (gfx1250) — matrix operand order repaired and checked by reading, not compiled, not run on hardware
//
#include <hip/hip_runtime.h>

typedef _Float16 v16h __attribute__((ext_vector_type(16)));
typedef _Float16 v8h  __attribute__((ext_vector_type(8)));
typedef float    v8f  __attribute__((ext_vector_type(8)));
typedef unsigned int u32x4 __attribute__((ext_vector_type(4)));
typedef int          i32x8 __attribute__((ext_vector_type(8)));
typedef int          i32x4 __attribute__((ext_vector_type(4)));

#define D_NODE 128
#define D_IN   384
#define D_HID  256
#define D_OUT  128

// LDS layout (dynamic): [0, 96KB) shared weight buffer (B1 pass0 / B1 pass1 / B2),
//                       [96KB, 160KB) per-wave 8KB h/out staging.
#define WBUF_BYTES   (96 * 1024)
#define STAGE_BYTES  8192
#define SMEM_BYTES   (WBUF_BYTES + 8 * STAGE_BYTES)

// ------------------------------------------------------------------
// Weight pre-swizzle into f16 WMMA B-operand blocks (32 lanes x 16 halfs
// = 1KB per 16x32 B tile). B layout for v_wmma_f32_16x16x32_f16:
// lane l -> column N = (l&15); half j -> K = ((j>=8)?16:0) + ((l>=16)?8:0) + (j&7).
// B1 block order: ((pass*12 + kt)*8 + nt) so each pass is one contiguous
// 96KB chunk (single TDM copy). B2 block order: (kt*8 + nt), 64KB.
// ------------------------------------------------------------------
__global__ void prep_w1_kernel(const float* __restrict__ W1, _Float16* __restrict__ B1) {
    int tid = blockIdx.x * blockDim.x + threadIdx.x;     // 2*12*8*512 total
    if (tid >= 2 * 12 * 8 * 512) return;
    int j   = tid & 15;
    int l   = (tid >> 4) & 31;
    int blk = tid >> 9;            // 0..191
    int nt  = blk & 7;
    int kt  = (blk % 96) >> 3;
    int p   = blk / 96;
    int k = kt * 32 + ((j >> 3) << 4) + ((l >> 4) << 3) + (j & 7);
    int n = (p * 8 + nt) * 16 + (l & 15);
    B1[tid] = (_Float16)W1[k * D_HID + n];
}

__global__ void prep_w2_kernel(const float* __restrict__ W2, _Float16* __restrict__ B2) {
    int tid = blockIdx.x * blockDim.x + threadIdx.x;     // 8*8*512 total
    if (tid >= 8 * 8 * 512) return;
    int j   = tid & 15;
    int l   = (tid >> 4) & 31;
    int blk = tid >> 9;
    int nt  = blk & 7;
    int kt  = blk >> 3;
    int k = kt * 32 + ((j >> 3) << 4) + ((l >> 4) << 3) + (j & 7);
    int n = nt * 16 + (l & 15);
    B2[tid] = (_Float16)W2[k * D_OUT + n];
}

// ------------------------------------------------------------------
// TDM 1-D copy: nq qwords from global gsrc -> LDS byte offset lds_off.
// D# per CDNA5 ISA: group0 {count=1, lds_addr,
// global_addr, type=2}; group1 {data_size=8B, tensor_dim0=tile_dim0=nq,
// tensor_dim1=tile_dim1=1}. Groups 2/3 zero (<=2D tensor).
// 6-arg builtin form (clang-23 / therock-10.0 headers).
// ------------------------------------------------------------------
__device__ __forceinline__ void tdm_copy_1d(const void* gsrc, unsigned lds_off, unsigned nq) {
    unsigned long long ga = (unsigned long long)(size_t)gsrc;
    u32x4 g0;
    g0[0] = 1u;                                            // count=1
    g0[1] = lds_off;                                       // lds_addr bytes
    g0[2] = (unsigned)(ga & 0xFFFFFFFFu);                  // global_addr[31:0]
    g0[3] = (unsigned)((ga >> 32) & 0x01FFFFFFu) | 0x80000000u;  // addr[56:32] | type=2
    i32x8 g1;
    g1[0] = (int)(3u << 16);                               // wg_mask=0, data_size=8B
    g1[1] = (int)((nq & 0xFFFFu) << 16);                   // tensor_dim0 lo16
    g1[2] = (int)(((nq >> 16) & 0xFFFFu) | (1u << 16));    // dim0 hi16 | tensor_dim1=1
    g1[3] = (int)((nq & 0xFFFFu) << 16);                   // dim1 hi=0 | tile_dim0=nq
    g1[4] = 1;                                             // tile_dim1=1, tile_dim2=0
    g1[5] = (int)nq;                                       // tensor_dim0_stride lo32
    g1[6] = 0;
    g1[7] = 0;
    i32x4 gz4 = {0, 0, 0, 0};
    i32x8 gz8 = {0, 0, 0, 0, 0, 0, 0, 0};
    __builtin_amdgcn_tensor_load_to_lds(g0, g1, gz4, gz4, gz8, 0);
}

// ------------------------------------------------------------------
// Main fused kernel: 8 waves / block, 1 tile of 16 edges per wave.
// Weights staged in LDS per block (TDM), shared by all waves.
// ------------------------------------------------------------------
__global__ __launch_bounds__(256)
void edge_mlp_kernel(const float* __restrict__ node_attr,
                     const float* __restrict__ edge_attr,
                     const int*   __restrict__ eidx,
                     const _Float16* __restrict__ B1,
                     const _Float16* __restrict__ B2,
                     const float* __restrict__ b1,
                     const float* __restrict__ b2,
                     const float* __restrict__ gamma,
                     const float* __restrict__ beta,
                     float* __restrict__ out,
                     int E) {
    extern __shared__ __align__(32) char smem[];
    _Float16* wbuf = (_Float16*)smem;                      // 96KB shared weights
    const int lane  = threadIdx.x & 31;
    const int wave  = threadIdx.x >> 5;
    const int tiles = (E + 15) >> 4;
    int tile  = blockIdx.x * 8 + wave;
    const bool valid = tile < tiles;
    if (!valid) tile = tiles - 1;                          // clamp; keep all waves at barriers

    const int lrow  = lane & 15;                           // A row owned by lane
    const int khalf = lane >> 4;                           // K-half of wave
    const int kbase = khalf << 3;                          // +8 K offset (A layout)

    int row  = tile * 16 + lrow;
    int rowc = row < E ? row : E - 1;
    int sidx = eidx[rowc];
    int ridx = eidx[E + rowc];
    const float* seg0 = node_attr + (long)sidx * D_NODE;   // sender   K 0..127
    const float* seg1 = node_attr + (long)ridx * D_NODE;   // receiver K 128..255
    const float* seg2 = edge_attr + (long)rowc * D_NODE;   // edge     K 256..383

    _Float16* hbuf = (_Float16*)(smem + WBUF_BYTES + wave * STAGE_BYTES); // 16x256 f16
    float*    obuf = (float*)   (smem + WBUF_BYTES + wave * STAGE_BYTES); // 16x128 f32

    // ================= GEMM1: [16x384] x [384x256] + b1, ReLU ===============
    #pragma unroll 1
    for (int pass = 0; pass < 2; ++pass) {
        // stage this pass's 96KB of B1 into LDS (wave 0 drives the TDM)
        if (wave == 0) {
            tdm_copy_1d(B1 + (size_t)pass * (96 * 512), 0, WBUF_BYTES / 8);
            __builtin_amdgcn_s_wait_tensorcnt(0);
        }
        __syncthreads();

        v8f acc[8];
        #pragma unroll
        for (int nt = 0; nt < 8; ++nt) {
            float bv = b1[(pass * 8 + nt) * 16 + lrow];
            #pragma unroll
            for (int i = 0; i < 8; ++i) acc[nt][i] = bv;
        }

        // prime A pipeline (kt = 0)
        float4 r0, r1, r2, r3;
        {
            const float* p = seg0 + kbase;
            r0 = *(const float4*)(p);      r1 = *(const float4*)(p + 4);
            r2 = *(const float4*)(p + 16); r3 = *(const float4*)(p + 20);
        }
        #pragma unroll
        for (int kt = 0; kt < 12; ++kt) {
            float4 n0 = r0, n1 = r1, n2 = r2, n3 = r3;
            if (kt < 11) {                 // prefetch next A tile (raw f32)
                int kn = kt + 1;
                const float* sp = (kn < 4) ? seg0 : (kn < 8) ? seg1 : seg2;
                const float* p  = sp + (kn & 3) * 32 + kbase;
                n0 = *(const float4*)(p);      n1 = *(const float4*)(p + 4);
                n2 = *(const float4*)(p + 16); n3 = *(const float4*)(p + 20);
            }
            v16h a;
            a[0]  = (_Float16)r0.x; a[1]  = (_Float16)r0.y;
            a[2]  = (_Float16)r0.z; a[3]  = (_Float16)r0.w;
            a[4]  = (_Float16)r1.x; a[5]  = (_Float16)r1.y;
            a[6]  = (_Float16)r1.z; a[7]  = (_Float16)r1.w;
            a[8]  = (_Float16)r2.x; a[9]  = (_Float16)r2.y;
            a[10] = (_Float16)r2.z; a[11] = (_Float16)r2.w;
            a[12] = (_Float16)r3.x; a[13] = (_Float16)r3.y;
            a[14] = (_Float16)r3.z; a[15] = (_Float16)r3.w;

            const _Float16* bb = wbuf + (size_t)(kt * 8) * 512 + lane * 16;
            v16h bcur = *(const v16h*)(bb);
            #pragma unroll
            for (int nt = 0; nt < 8; ++nt) {
                v16h bn = bcur;
                if (nt < 7) bn = *(const v16h*)(bb + (nt + 1) * 512);  // prefetch next B
                acc[nt] = __builtin_amdgcn_wmma_f32_16x16x32_f16(
                    false, a, false, bcur, (short)0, acc[nt], false, false);
                bcur = bn;
            }
            r0 = n0; r1 = n1; r2 = n2; r3 = n3;
        }

        // ReLU + scatter D-layout -> row-major f16 (per-wave LDS)
        #pragma unroll
        for (int nt = 0; nt < 8; ++nt) {
            int col = (pass * 8 + nt) * 16 + lrow;
            #pragma unroll
            for (int rr = 0; rr < 8; ++rr) {
                float v = acc[nt][rr];
                v = v > 0.f ? v : 0.f;
                hbuf[(kbase + rr) * D_HID + col] = (_Float16)v;
            }
        }
        __syncthreads();                   // all waves done with this wbuf pass
    }

    // ================= stage B2 (64KB) =================
    if (wave == 0) {
        tdm_copy_1d(B2, 0, (64 * 1024) / 8);
        __builtin_amdgcn_s_wait_tensorcnt(0);
    }
    __syncthreads();
    asm volatile("" ::: "memory");

    // ================= GEMM2: [16x256] x [256x128] + b2 =================
    v8f acc2[8];
    #pragma unroll
    for (int nt = 0; nt < 8; ++nt) {
        float bv = b2[nt * 16 + lrow];
        #pragma unroll
        for (int i = 0; i < 8; ++i) acc2[nt][i] = bv;
    }
    #pragma unroll
    for (int kt = 0; kt < 8; ++kt) {
        const _Float16* ap = hbuf + lrow * D_HID + kt * 32 + kbase;
        v8h lo = *(const v8h*)(ap);        // K = base+0..7
        v8h hi = *(const v8h*)(ap + 16);   // K = base+16..23
        v16h a = __builtin_shufflevector(lo, hi,
                    0,1,2,3,4,5,6,7,8,9,10,11,12,13,14,15);
        const _Float16* bb = wbuf + (size_t)(kt * 8) * 512 + lane * 16;
        v16h bcur = *(const v16h*)(bb);
        #pragma unroll
        for (int nt = 0; nt < 8; ++nt) {
            v16h bn = bcur;
            if (nt < 7) bn = *(const v16h*)(bb + (nt + 1) * 512);
            acc2[nt] = __builtin_amdgcn_wmma_f32_16x16x32_f16(
                false, a, false, bcur, (short)0, acc2[nt], false, false);
            bcur = bn;
        }
    }
    asm volatile("" ::: "memory");

    // stage 16x128 f32 tile in per-wave LDS (same-wave DS ops are in-order)
    #pragma unroll
    for (int nt = 0; nt < 8; ++nt) {
        int col = nt * 16 + lrow;
        #pragma unroll
        for (int rr = 0; rr < 8; ++rr)
            obuf[(kbase + rr) * D_OUT + col] = acc2[nt][rr];
    }
    asm volatile("" ::: "memory");

    // ================= LayerNorm(128): 2 lanes per row =================
    const float* rp = obuf + lrow * D_OUT + khalf * 64;
    float sum = 0.f, sq = 0.f;
    #pragma unroll
    for (int i = 0; i < 64; ++i) { float v = rp[i]; sum += v; sq += v * v; }
    sum += __shfl_xor(sum, 16, 32);
    sq  += __shfl_xor(sq,  16, 32);
    float mean = sum * (1.0f / 128.0f);
    float var  = sq  * (1.0f / 128.0f) - mean * mean;
    float rstd = rsqrtf(var + 1e-5f);
    if (valid && row < E) {
        float* op = out + (long)row * D_OUT + khalf * 64;
        #pragma unroll
        for (int i = 0; i < 64; ++i) {
            int c = khalf * 64 + i;
            op[i] = (rp[i] - mean) * rstd * gamma[c] + beta[c];
        }
    }
}

// ------------------------------------------------------------------
// Launch
// ------------------------------------------------------------------
extern "C" void kernel_launch(void* const* d_in, const int* in_sizes, int n_in,
                              void* d_out, int out_size, void* d_ws, size_t ws_size,
                              hipStream_t stream) {
    const float* node_attr = (const float*)d_in[0];
    const float* edge_attr = (const float*)d_in[1];
    const int*   eidx      = (const int*)  d_in[2];
    const float* W1        = (const float*)d_in[3];
    const float* b1        = (const float*)d_in[4];
    const float* W2        = (const float*)d_in[5];
    const float* b2        = (const float*)d_in[6];
    const float* gamma     = (const float*)d_in[7];
    const float* beta      = (const float*)d_in[8];
    float* out = (float*)d_out;

    int E = in_sizes[2] / 2;   // edge_contact_index is [2, E]

    _Float16* B1ws = (_Float16*)d_ws;          // 192 KB (two 96KB pass chunks)
    _Float16* B2ws = B1ws + 2 * 12 * 8 * 512;  //  64 KB

    prep_w1_kernel<<<(2 * 12 * 8 * 512 + 255) / 256, 256, 0, stream>>>(W1, B1ws);
    prep_w2_kernel<<<(8 * 8 * 512 + 255) / 256, 256, 0, stream>>>(W2, B2ws);

    int tiles  = (E + 15) / 16;
    int blocks = (tiles + 7) / 8;
    edge_mlp_kernel<<<blocks, 256, SMEM_BYTES, stream>>>(
        node_attr, edge_attr, eidx, B1ws, B2ws, b1, b2, gamma, beta, out, E);
}
